// PointNetSetAbstraction_19061064860312
// MI455X (gfx1250) — matrix operand order repaired and checked by reading, not compiled, not run on hardware
//
#include <hip/hip_runtime.h>

typedef __attribute__((ext_vector_type(16))) _Float16 v16h;
typedef __attribute__((ext_vector_type(8)))  _Float16 v8h;
typedef __attribute__((ext_vector_type(8)))  float    v8f;

#define BB   16
#define NN   4096
#define CC   64
#define SS   1024
#define KK   32
#define MM   (BB * SS * KK)   // 524288 rows in the MLP GEMMs
#define KPAD 96               // 67 padded up to multiple of 32

// ---------------------------------------------------------------------------
// 0. zero the BN-statistics scratch (harness does not re-zero between replays)
// ---------------------------------------------------------------------------
__global__ void zero_stats_kernel(float* __restrict__ p, int n) {
    int i = blockIdx.x * 256 + threadIdx.x;
    if (i < n) p[i] = 0.0f;
}

// ---------------------------------------------------------------------------
// 1. Farthest point sampling: one block per batch, serial 1024-step argmax.
//    xyz cached in LDS (SoA), distance array in LDS, tree argmax reduction.
// ---------------------------------------------------------------------------
__global__ __launch_bounds__(1024) void fps_kernel(const float* __restrict__ xyz,
                                                   int* __restrict__ fps_idx,
                                                   float* __restrict__ new_xyz) {
    const int b = blockIdx.x;
    const int tid = threadIdx.x;
    __shared__ float px[NN], py[NN], pz[NN];
    __shared__ float dist[NN];
    __shared__ float rv[1024];
    __shared__ int   ri[1024];

    const float* bx = xyz + (size_t)b * NN * 3;
    for (int i = tid; i < NN; i += 1024) {
        px[i] = bx[i * 3 + 0];
        py[i] = bx[i * 3 + 1];
        pz[i] = bx[i * 3 + 2];
        dist[i] = 1e10f;
    }
    __syncthreads();

    int far = 0;
    for (int it = 0; it < SS; ++it) {
        if (tid == 0) {
            fps_idx[b * SS + it] = far;
            new_xyz[((size_t)b * SS + it) * 3 + 0] = px[far];
            new_xyz[((size_t)b * SS + it) * 3 + 1] = py[far];
            new_xyz[((size_t)b * SS + it) * 3 + 2] = pz[far];
        }
        const float cx = px[far], cy = py[far], cz = pz[far];
        float bestv = -1.0f;
        int   besti = 0;
        #pragma unroll
        for (int r = 0; r < NN / 1024; ++r) {
            const int i = tid + r * 1024;
            const float dx = px[i] - cx, dy = py[i] - cy, dz = pz[i] - cz;
            const float d = dx * dx + dy * dy + dz * dz;
            const float nd = fminf(dist[i], d);
            dist[i] = nd;
            if (nd > bestv) { bestv = nd; besti = i; }   // i ascending: first-index tie-break
        }
        rv[tid] = bestv; ri[tid] = besti;
        __syncthreads();
        for (int s = 512; s > 0; s >>= 1) {
            if (tid < s) {
                const float ov = rv[tid + s];
                const int   oi = ri[tid + s];
                if (ov > rv[tid] || (ov == rv[tid] && oi < ri[tid])) { rv[tid] = ov; ri[tid] = oi; }
            }
            __syncthreads();
        }
        far = ri[0];
        __syncthreads();
    }
}

// ---------------------------------------------------------------------------
// 2. kNN (32 smallest sq-dists) + gather + concat -> X0 f16 [M, 96] (zero pad)
//    One block of 256 threads per query point; distances in LDS; 32 serial
//    argmin selection passes (matches top_k semantics, index tie-break).
// ---------------------------------------------------------------------------
__global__ __launch_bounds__(256) void knn_gather_kernel(const float* __restrict__ xyz,
                                                         const float* __restrict__ points,
                                                         const float* __restrict__ new_xyz,
                                                         _Float16* __restrict__ X0) {
    const int q   = blockIdx.x;       // 0 .. B*S-1
    const int b   = q >> 10;
    const int tid = threadIdx.x;
    __shared__ float d[NN];
    __shared__ float rv[256];
    __shared__ int   ri[256];
    __shared__ int   sel[KK];

    const float qx = new_xyz[q * 3 + 0];
    const float qy = new_xyz[q * 3 + 1];
    const float qz = new_xyz[q * 3 + 2];
    const float* bx = xyz + (size_t)b * NN * 3;

    for (int i = tid; i < NN; i += 256) {
        const float dx = bx[i * 3 + 0] - qx;
        const float dy = bx[i * 3 + 1] - qy;
        const float dz = bx[i * 3 + 2] - qz;
        d[i] = dx * dx + dy * dy + dz * dz;
    }
    __syncthreads();

    for (int j = 0; j < KK; ++j) {
        float bv = 3.0e38f;
        int   bi = 0;
        #pragma unroll
        for (int r = 0; r < NN / 256; ++r) {
            const int i = tid + r * 256;
            const float v = d[i];
            if (v < bv || (v == bv && i < bi)) { bv = v; bi = i; }
        }
        rv[tid] = bv; ri[tid] = bi;
        __syncthreads();
        for (int s = 128; s > 0; s >>= 1) {
            if (tid < s) {
                const float ov = rv[tid + s];
                const int   oi = ri[tid + s];
                if (ov < rv[tid] || (ov == rv[tid] && oi < ri[tid])) { rv[tid] = ov; ri[tid] = oi; }
            }
            __syncthreads();
        }
        if (tid == 0) { sel[j] = ri[0]; d[ri[0]] = 3.9e38f; }
        __syncthreads();
    }

    const float* bp = points + (size_t)b * NN * CC;
    for (int t = tid; t < KK * KPAD; t += 256) {
        const int j = t / KPAD;
        const int c = t - j * KPAD;
        const int idx = sel[j];
        float v;
        if (c < 3)       v = bx[idx * 3 + c] - (c == 0 ? qx : (c == 1 ? qy : qz));
        else if (c < 67) v = bp[(size_t)idx * CC + (c - 3)];
        else             v = 0.0f;
        X0[((size_t)q * KK + j) * KPAD + c] = (_Float16)v;
    }
}

// ---------------------------------------------------------------------------
// 3. WMMA GEMM:  C[M,O] = A[M,Ka](f16) x W[O,Kw](f32->f16) + bias
//    8 waves/block, each wave one 16x16 tile; block tile = 128 rows x 16 cols.
//    Epilogue: store f16 pre-BN activation + accumulate per-channel sum/sumsq
//    (shfl + LDS block reduction + one atomic per channel per block).
// ---------------------------------------------------------------------------
__global__ __launch_bounds__(256) void gemm_bn_kernel(const _Float16* __restrict__ A, int Ka,
                                                      const float* __restrict__ W, int Kw, int O,
                                                      const float* __restrict__ bias,
                                                      _Float16* __restrict__ Cout,
                                                      float* __restrict__ gsum,
                                                      float* __restrict__ gsumsq) {
    const int tid     = threadIdx.x;
    const int lane    = tid & 31;
    const int wv      = tid >> 5;
    const int rowTile = blockIdx.x * 128;
    const int colBase = blockIdx.y * 16;

    __shared__ __attribute__((aligned(32))) _Float16 Wlds[16 * KPAD];
    __shared__ float sred[8][16];
    __shared__ float qred[8][16];

    // Stage W^T column tile (16 output channels x Ka) in LDS as f16, zero-pad k>=Kw.
    for (int t = tid; t < 16 * Ka; t += 256) {
        const int n = t / Ka;
        const int k = t - n * Ka;
        Wlds[n * Ka + k] = (k < Kw) ? (_Float16)W[(size_t)(colBase + n) * Kw + k] : (_Float16)0.0f;
    }
    __syncthreads();

    const int half = lane >> 4;      // K-half selector
    const int l15  = lane & 15;      // M for A-frag, N for B/C-frag
    const int row  = rowTile + wv * 16 + l15;
    const _Float16* Arow = A + (size_t)row * Ka;

    v8f c = {};
    for (int kk = 0; kk < Ka; kk += 32) {
        // A-fragment: lane l15 = row, elements 0..7 = K kk+half*8.., 8..15 = K kk+16+half*8..
        v8h alo = *(const v8h*)(Arow + kk + half * 8);
        v8h ahi = *(const v8h*)(Arow + kk + 16 + half * 8);
        v16h a = __builtin_shufflevector(alo, ahi, 0, 1, 2, 3, 4, 5, 6, 7,
                                                   8, 9, 10, 11, 12, 13, 14, 15);
        // B-fragment: lane l15 = column N, elements 0..7 = K kk+half*8.., 8..15 = K kk+16+half*8..
        v8h blo = *(const v8h*)(&Wlds[l15 * Ka + kk + half * 8]);
        v8h bhi = *(const v8h*)(&Wlds[l15 * Ka + kk + 16 + half * 8]);
        v16h bm = __builtin_shufflevector(blo, bhi, 0, 1, 2, 3, 4, 5, 6, 7,
                                                    8, 9, 10, 11, 12, 13, 14, 15);
        c = __builtin_amdgcn_wmma_f32_16x16x32_f16(false, a, false, bm, (short)0, c, false, false);
    }

    const float bv = bias[colBase + l15];
    float s = 0.0f, qq = 0.0f;
    #pragma unroll
    for (int i = 0; i < 8; ++i) {
        const float v = c[i] + bv;                      // pre-BN activation
        s  += v;
        qq += v * v;
        const int r = rowTile + wv * 16 + half * 8 + i; // C layout: VGPR i -> M = i + 8*half
        Cout[(size_t)r * O + colBase + l15] = (_Float16)v;
    }

    // lanes L and L^16 hold the same channel: fold, then block-reduce in LDS.
    s  += __shfl_xor(s, 16);
    qq += __shfl_xor(qq, 16);
    if (half == 0) { sred[wv][l15] = s; qred[wv][l15] = qq; }
    __syncthreads();
    if (tid < 16) {
        float ts = 0.0f, tq = 0.0f;
        #pragma unroll
        for (int w = 0; w < 8; ++w) { ts += sred[w][tid]; tq += qred[w][tid]; }
        atomicAdd(&gsum[colBase + tid], ts);
        atomicAdd(&gsumsq[colBase + tid], tq);
    }
}

// ---------------------------------------------------------------------------
// 4. BN statistics finalize: mean / invstd per channel
// ---------------------------------------------------------------------------
__global__ void bn_stats_finalize(const float* __restrict__ gsum,
                                  const float* __restrict__ gsumsq,
                                  float* __restrict__ mean,
                                  float* __restrict__ invstd,
                                  int O, float invM) {
    const int o = threadIdx.x;
    if (o < O) {
        const float m = gsum[o] * invM;
        const float v = gsumsq[o] * invM - m * m;
        mean[o]   = m;
        invstd[o] = rsqrtf(v + 1e-5f);
    }
}

// ---------------------------------------------------------------------------
// 5. In-place BN + ReLU on an f16 activation buffer
// ---------------------------------------------------------------------------
__global__ __launch_bounds__(256) void bn_relu_kernel(_Float16* __restrict__ h, int omask,
                                                      const float* __restrict__ mean,
                                                      const float* __restrict__ invstd,
                                                      const float* __restrict__ g,
                                                      const float* __restrict__ be,
                                                      size_t total) {
    const size_t i = (size_t)blockIdx.x * 256 + threadIdx.x;
    if (i >= total) return;
    const int o = (int)(i & (size_t)omask);
    const float v = (float)h[i];
    const float r = (v - mean[o]) * invstd[o] * g[o] + be[o];
    h[i] = (_Float16)fmaxf(r, 0.0f);
}

// ---------------------------------------------------------------------------
// 6. Fused BN + ReLU + max-pool over the 32 neighbors (layer 3 epilogue)
// ---------------------------------------------------------------------------
__global__ __launch_bounds__(256) void maxpool_kernel(const _Float16* __restrict__ h,
                                                      const float* __restrict__ mean,
                                                      const float* __restrict__ invstd,
                                                      const float* __restrict__ g,
                                                      const float* __restrict__ be,
                                                      float* __restrict__ out) {
    const int i = blockIdx.x * 256 + threadIdx.x;   // 0 .. B*S*128-1
    const int o = i & 127;
    const int q = i >> 7;
    const float m = mean[o], is = invstd[o], gg = g[o], bb = be[o];
    float best = 0.0f;                               // ReLU outputs are >= 0
    #pragma unroll
    for (int j = 0; j < KK; ++j) {
        const float v = (float)h[((size_t)q * KK + j) * 128 + o];
        float r = (v - m) * is * gg + bb;
        r = fmaxf(r, 0.0f);
        best = fmaxf(best, r);
    }
    out[i] = best;
}

// ---------------------------------------------------------------------------
// launch
// ---------------------------------------------------------------------------
extern "C" void kernel_launch(void* const* d_in, const int* in_sizes, int n_in,
                              void* d_out, int out_size, void* d_ws, size_t ws_size,
                              hipStream_t stream) {
    const float* xyz    = (const float*)d_in[0];
    const float* points = (const float*)d_in[1];
    const float* w1 = (const float*)d_in[2];
    const float* b1 = (const float*)d_in[3];
    const float* g1 = (const float*)d_in[4];
    const float* be1 = (const float*)d_in[5];
    const float* w2 = (const float*)d_in[6];
    const float* b2 = (const float*)d_in[7];
    const float* g2 = (const float*)d_in[8];
    const float* be2 = (const float*)d_in[9];
    const float* w3 = (const float*)d_in[10];
    const float* b3 = (const float*)d_in[11];
    const float* g3 = (const float*)d_in[12];
    const float* be3 = (const float*)d_in[13];

    float* out      = (float*)d_out;
    float* new_xyz  = out;                       // [B,S,3] flat
    float* out_feat = out + (size_t)BB * SS * 3; // [B,S,128] flat

    // workspace carve (bytes)
    char* ws = (char*)d_ws;
    float* stats = (float*)ws;                   // 3 layers x (sum[128], sumsq[128]) = 768 f32
    float* meanv = stats + 768;                  // 3 layers x (mean[128], invstd[128]) = 768 f32
    int*   fpsi  = (int*)(ws + 8192);            // 16384 ints
    _Float16* bufA = (_Float16*)(ws + 8192 + 65536);                       // M x 96 f16
    _Float16* bufB = (_Float16*)(ws + 8192 + 65536 + (size_t)MM * KPAD * 2); // M x 128 f16

    const float invM = 1.0f / (float)MM;

    zero_stats_kernel<<<(1536 + 255) / 256, 256, 0, stream>>>(stats, 1536);
    fps_kernel<<<BB, 1024, 0, stream>>>(xyz, fpsi, new_xyz);
    knn_gather_kernel<<<BB * SS, 256, 0, stream>>>(xyz, points, new_xyz, bufA);

    // layer 1: [M,96(67)] x [67,64]
    gemm_bn_kernel<<<dim3(MM / 128, 64 / 16), 256, 0, stream>>>(
        bufA, KPAD, w1, 67, 64, b1, bufB, stats + 0, stats + 128);
    bn_stats_finalize<<<1, 128, 0, stream>>>(stats + 0, stats + 128, meanv + 0, meanv + 128, 64, invM);
    bn_relu_kernel<<<(unsigned)(((size_t)MM * 64) / 256), 256, 0, stream>>>(
        bufB, 63, meanv + 0, meanv + 128, g1, be1, (size_t)MM * 64);

    // layer 2: [M,64] x [64,64]
    gemm_bn_kernel<<<dim3(MM / 128, 64 / 16), 256, 0, stream>>>(
        bufB, 64, w2, 64, 64, b2, bufA, stats + 256, stats + 384);
    bn_stats_finalize<<<1, 128, 0, stream>>>(stats + 256, stats + 384, meanv + 256, meanv + 384, 64, invM);
    bn_relu_kernel<<<(unsigned)(((size_t)MM * 64) / 256), 256, 0, stream>>>(
        bufA, 63, meanv + 256, meanv + 384, g2, be2, (size_t)MM * 64);

    // layer 3: [M,64] x [64,128]
    gemm_bn_kernel<<<dim3(MM / 128, 128 / 16), 256, 0, stream>>>(
        bufA, 64, w3, 64, 128, b3, bufB, stats + 512, stats + 640);
    bn_stats_finalize<<<1, 128, 0, stream>>>(stats + 512, stats + 640, meanv + 512, meanv + 640, 128, invM);

    maxpool_kernel<<<(BB * SS * 128) / 256, 256, 0, stream>>>(
        bufB, meanv + 512, meanv + 640, g3, be3, out_feat);
}
